// LSTMOtm_61933428408459
// MI455X (gfx1250) — matrix operand order repaired and checked by reading, not compiled, not run on hardware
//
#include <hip/hip_runtime.h>
#include <hip/hip_bf16.h>
#include <stdint.h>

// ---------------------------------------------------------------------------
// Problem constants
// ---------------------------------------------------------------------------
#define BB    64          // batch
#define TCAP  32          // caption length cap
#define TDEC  31          // decode steps (T_CAP - 1)
#define VV    32000       // vocab
#define EE    1024        // encoder dim
#define DD    1024        // hidden dim
#define G4    4096        // 4*D (gate rows)
#define KC    2048        // E + D  (combined K for gates GEMM)
#define MROWS (TDEC * BB) // 1984 rows of the batched FC GEMM (62 tiles of 32)

// ---------------------------------------------------------------------------
// WMMA vector types (wave32, gfx1250)
// ---------------------------------------------------------------------------
typedef __attribute__((ext_vector_type(16))) __bf16         v16bf;
typedef __attribute__((ext_vector_type(16))) unsigned short v16u;
typedef __attribute__((ext_vector_type(8)))  unsigned short u16x8;
typedef __attribute__((ext_vector_type(8)))  float          v8f;

union BFPack { v16u u; v16bf b; };
union V8F    { v8f v; float f[8]; };

__device__ __forceinline__ unsigned short f2bf(float f) {
    unsigned int u = __float_as_uint(f);
    u += 0x7fffu + ((u >> 16) & 1u);       // round-to-nearest-even
    return (unsigned short)(u >> 16);
}

__device__ __forceinline__ float sigmoidf_(float x) {
    return 1.0f / (1.0f + __expf(-x));
}

// A fragment: 16x32 bf16 tile from an LDS buffer with row stride `stride`.
// lane<16 : K = kb + {0..15};  lane>=16 : K = kb + {16..31}
__device__ __forceinline__ v16bf load_a_frag(const unsigned short* At, int row,
                                             int stride, int kb, int kh) {
    u16x8 lo = *(const u16x8*)(At + row * stride + kb + kh * 16);
    u16x8 hi = *(const u16x8*)(At + row * stride + kb + kh * 16 + 8);
    BFPack t; t.u.lo = lo; t.u.hi = hi;
    return t.b;
}

// B fragment: 32x16 (KxN). Each lane holds one output column n, 16 consecutive
// K values; lane-half selects K 0..15 vs 16..31. W stored row-major [N][K].
__device__ __forceinline__ v16bf load_b_frag(const unsigned short* __restrict__ W,
                                             size_t off) {
    u16x8 lo = *(const u16x8*)(W + off);
    u16x8 hi = *(const u16x8*)(W + off + 8);
    BFPack t; t.u.lo = lo; t.u.hi = hi;
    return t.b;
}

// ---------------------------------------------------------------------------
// 1) prep: stable descending argsort of lengths (tiny), writes output tail
// ---------------------------------------------------------------------------
__global__ void prep_kernel(const int* __restrict__ lengths, // [B,1]
                            int* __restrict__ sind,
                            int* __restrict__ declen,
                            float* __restrict__ out_tail) {   // 128 floats
    if (threadIdx.x != 0 || blockIdx.x != 0) return;
    bool used[BB];
    for (int i = 0; i < BB; ++i) used[i] = false;
    for (int i = 0; i < BB; ++i) {
        int best = -1; long long bl = -1;
        for (int j = 0; j < BB; ++j) {
            if (!used[j] && (long long)lengths[j] > bl) { bl = lengths[j]; best = j; }
        }
        used[best] = true;
        sind[i]   = best;
        declen[i] = (int)bl - 1;
        out_tail[i]      = (float)((int)bl - 1); // decode_lengths
        out_tail[BB + i] = (float)best;          // sort_ind
    }
}

// ---------------------------------------------------------------------------
// 2) conversions to bf16
// ---------------------------------------------------------------------------
__global__ void convert_wc_kernel(const float* __restrict__ Wih,
                                  const float* __restrict__ Whh,
                                  unsigned short* __restrict__ Wc) { // [4096][2048]
    int i = blockIdx.x * blockDim.x + threadIdx.x;     // 0 .. 4096*2048-1
    int r = i >> 11, k = i & 2047;
    float v = (k < DD) ? Wih[r * DD + k] : Whh[r * DD + (k - DD)];
    Wc[i] = f2bf(v);
}

__global__ void bias_sum_kernel(const float* __restrict__ bih,
                                const float* __restrict__ bhh,
                                float* __restrict__ bsum) {
    int i = blockIdx.x * blockDim.x + threadIdx.x;     // 0..4095
    if (i < G4) bsum[i] = bih[i] + bhh[i];
}

__global__ void convert_fcw_kernel(const float* __restrict__ fcW,
                                   unsigned short* __restrict__ fcWb) {
    size_t i = (size_t)blockIdx.x * blockDim.x + threadIdx.x; // 0..32768000-1
    fcWb[i] = f2bf(fcW[i]);
}

__global__ void zero_hc_kernel(float* __restrict__ h, float* __restrict__ c) {
    int i = blockIdx.x * blockDim.x + threadIdx.x;     // 0 .. 64*1024-1
    h[i] = 0.0f; c[i] = 0.0f;
}

// ---------------------------------------------------------------------------
// 3a) gates GEMM:  gates(64x4096) = [x_t || h](64x2048) @ Wc^T + bsum
//     grid = 64 blocks (4 M-tiles x 16 N-chunks of 256), 256 threads (8 waves)
// ---------------------------------------------------------------------------
__global__ void gates_gemm_kernel(const float* __restrict__ enc,
                                  const float* __restrict__ emb,
                                  const int*  __restrict__ caps,
                                  const unsigned short* __restrict__ Wc,
                                  const float* __restrict__ bsum,
                                  const float* __restrict__ h,
                                  float* __restrict__ gates,
                                  const int* __restrict__ sind,
                                  int step) {
    __shared__ unsigned short At[16 * 1024];  // 32 KB: 16 rows x 1024 K (half)

    const int blk = blockIdx.x;
    const int m0  = (blk >> 4) * 16;          // M-tile base (0,16,32,48)
    const int n0  = (blk & 15) * 256;         // N-chunk base
    const int tid = threadIdx.x;
    const int lane = tid & 31, wave = tid >> 5;
    const int row = lane & 15, kh = lane >> 4;

    v8f acc0 = {}; v8f acc1 = {};

    for (int half = 0; half < 2; ++half) {
        const int kk0 = half * 1024;
        // --- cooperatively stage A half-tile, converting fp32 -> bf16 -------
        for (int i = tid; i < 16 * 1024; i += 256) {
            int r = i >> 10, k = i & 1023;
            int gk = kk0 + k;
            float v;
            if (gk < EE) {  // x_t portion
                int srow = sind[m0 + r];
                if (step == 0) {
                    v = enc[srow * EE + gk];
                } else {
                    int tok = caps[srow * TCAP + (step - 1)];
                    v = emb[(size_t)tok * EE + gk];
                }
            } else {        // h portion
                v = h[(m0 + r) * DD + (gk - EE)];
            }
            At[i] = f2bf(v);
        }
        __syncthreads();

        // --- WMMA over this K half -----------------------------------------
        const int nA = n0 + wave * 32 + (lane & 15);
        #pragma unroll 4
        for (int ks = 0; ks < 32; ++ks) {
            const int kb = ks * 32;
            v16bf a = load_a_frag(At, row, 1024, kb, kh);
            v16bf b0 = load_b_frag(Wc, (size_t)nA * KC + (kk0 + kb) + kh * 16);
            v16bf b1 = load_b_frag(Wc, (size_t)(nA + 16) * KC + (kk0 + kb) + kh * 16);
            acc0 = __builtin_amdgcn_wmma_f32_16x16x32_bf16(
                false, a, false, b0, (short)0, acc0, false, false);
            acc1 = __builtin_amdgcn_wmma_f32_16x16x32_bf16(
                false, a, false, b1, (short)0, acc1, false, false);
        }
        __syncthreads();
    }

    // --- epilogue: add bias, store fp32 gates -------------------------------
    #pragma unroll
    for (int tile = 0; tile < 2; ++tile) {
        V8F acc; acc.v = tile ? acc1 : acc0;
        int n = n0 + wave * 32 + tile * 16 + (lane & 15);
        float bs = bsum[n];
        #pragma unroll
        for (int vr = 0; vr < 8; ++vr) {
            int r = (lane < 16) ? vr : vr + 8;
            gates[(m0 + r) * G4 + n] = acc.f[vr] + bs;
        }
    }
}

// ---------------------------------------------------------------------------
// 3b) elementwise LSTM state update + masked h capture (bf16)
// ---------------------------------------------------------------------------
__global__ void lstm_update_kernel(float* __restrict__ h, float* __restrict__ c,
                                   const float* __restrict__ gates,
                                   const int* __restrict__ declen,
                                   unsigned short* __restrict__ Hbuf,
                                   float* __restrict__ maskbuf,
                                   int step) {
    int idx = blockIdx.x * blockDim.x + threadIdx.x;  // 0 .. 64*1024-1
    int b = idx >> 10, k = idx & 1023;
    const float* g = gates + b * G4;
    float gi = g[k], gf = g[DD + k], gg = g[2 * DD + k], go = g[3 * DD + k];
    float cprev = c[idx], hprev = h[idx];
    float cn = sigmoidf_(gf) * cprev + sigmoidf_(gi) * tanhf(gg);
    float hn = sigmoidf_(go) * tanhf(cn);
    if (step == 0) {
        h[idx] = hn; c[idx] = cn;
    } else {
        bool m = (step - 1) < declen[b];
        float hq = m ? hn : hprev;
        float cq = m ? cn : cprev;
        h[idx] = hq; c[idx] = cq;
        Hbuf[((size_t)(step - 1) * BB + b) * DD + k] = f2bf(hq);
        if (k == 0) maskbuf[(step - 1) * BB + b] = m ? 1.0f : 0.0f;
    }
}

// ---------------------------------------------------------------------------
// 4) batched FC GEMM: preds(1984 x 32000) = H(1984x1024) @ fcW^T + fc_b, masked
//    2M x 2N register tiling per wave: block covers 32 rows x 256 cols.
//    grid = 62 M-tiles x 125 N-chunks = 7750 blocks, 256 threads (8 waves).
//    Per k-step/wave: 4 LDS b128 + 4 global b128 feed 4 WMMAs (2 mem/wmma).
// ---------------------------------------------------------------------------
__global__ void fc_gemm_kernel(const unsigned short* __restrict__ Hbuf,
                               const unsigned short* __restrict__ fcWb,
                               const float* __restrict__ fcb,
                               const float* __restrict__ maskbuf,
                               float* __restrict__ out) {
    __shared__ unsigned short At[32 * 512];   // 32 KB: 32 rows x 512 K (half)

    const int mt = blockIdx.x / 125;
    const int nc = blockIdx.x % 125;
    const int m0 = mt * 32;
    const int n0 = nc * 256;
    const int tid = threadIdx.x;
    const int lane = tid & 31, wave = tid >> 5;
    const int row = lane & 15, kh = lane >> 4;

    v8f acc00 = {}; v8f acc01 = {}; v8f acc10 = {}; v8f acc11 = {};
    const int nA = n0 + wave * 32 + (lane & 15);

    for (int half = 0; half < 2; ++half) {
        const int kk0 = half * 512;
        // stage A: rows m0..m0+31, K slice kk0..kk0+511 (bf16 copy)
        for (int i = tid; i < 32 * 512; i += 256) {
            int r = i >> 9, k = i & 511;
            At[i] = Hbuf[(size_t)(m0 + r) * DD + kk0 + k];
        }
        __syncthreads();

        // prefetch next K-phase of B into cache while computing this one
        if (half == 0) {
            __builtin_prefetch(fcWb + (size_t)nA * DD + 512 + kh * 256, 0, 1);
            __builtin_prefetch(fcWb + (size_t)(nA + 16) * DD + 512 + kh * 256, 0, 1);
        }

        #pragma unroll 4
        for (int ks = 0; ks < 16; ++ks) {
            const int kb = ks * 32;
            v16bf a0 = load_a_frag(At, row,      512, kb, kh);
            v16bf a1 = load_a_frag(At, row + 16, 512, kb, kh);
            v16bf b0 = load_b_frag(fcWb, (size_t)nA * DD + (kk0 + kb) + kh * 16);
            v16bf b1 = load_b_frag(fcWb, (size_t)(nA + 16) * DD + (kk0 + kb) + kh * 16);
            acc00 = __builtin_amdgcn_wmma_f32_16x16x32_bf16(
                false, a0, false, b0, (short)0, acc00, false, false);
            acc01 = __builtin_amdgcn_wmma_f32_16x16x32_bf16(
                false, a0, false, b1, (short)0, acc01, false, false);
            acc10 = __builtin_amdgcn_wmma_f32_16x16x32_bf16(
                false, a1, false, b0, (short)0, acc10, false, false);
            acc11 = __builtin_amdgcn_wmma_f32_16x16x32_bf16(
                false, a1, false, b1, (short)0, acc11, false, false);
        }
        __syncthreads();
    }

    // epilogue: bias + mask, scatter into (B, TDEC, V) output
    #pragma unroll
    for (int mi = 0; mi < 2; ++mi) {
        #pragma unroll
        for (int ni = 0; ni < 2; ++ni) {
            V8F acc;
            acc.v = mi ? (ni ? acc11 : acc10) : (ni ? acc01 : acc00);
            int n = n0 + wave * 32 + ni * 16 + (lane & 15);
            float bias = fcb[n];
            #pragma unroll
            for (int vr = 0; vr < 8; ++vr) {
                int r = mi * 16 + ((lane < 16) ? vr : vr + 8);
                int M = m0 + r;                 // M = t*64 + b
                int t = M >> 6, b = M & 63;
                float msk = maskbuf[M];
                out[(size_t)b * TDEC * VV + (size_t)t * VV + n] =
                    msk * (acc.f[vr] + bias);
            }
        }
    }
}

// ---------------------------------------------------------------------------
// Host launcher
// ---------------------------------------------------------------------------
extern "C" void kernel_launch(void* const* d_in, const int* in_sizes, int n_in,
                              void* d_out, int out_size, void* d_ws, size_t ws_size,
                              hipStream_t stream) {
    const float* enc  = (const float*)d_in[0];   // (64,1024)
    const int*   caps = (const int*)d_in[1];     // (64,32)
    const int*   lens = (const int*)d_in[2];     // (64,1)
    const float* emb  = (const float*)d_in[3];   // (32000,1024)
    const float* Wih  = (const float*)d_in[4];   // (4096,1024)
    const float* Whh  = (const float*)d_in[5];   // (4096,1024)
    const float* bih  = (const float*)d_in[6];   // (4096,)
    const float* bhh  = (const float*)d_in[7];   // (4096,)
    const float* fcW  = (const float*)d_in[8];   // (32000,1024)
    const float* fcb  = (const float*)d_in[9];   // (32000,)
    float* out = (float*)d_out;

    // workspace carve-up (all offsets 256B aligned)
    char* ws = (char*)d_ws;
    size_t off = 0;
    unsigned short* Wc    = (unsigned short*)(ws + off); off += (size_t)G4 * KC * 2;      // 16 MB
    unsigned short* fcWb  = (unsigned short*)(ws + off); off += (size_t)VV * DD * 2;      // 64 MB
    float*          bsum  = (float*)(ws + off);          off += (size_t)G4 * 4;
    float*          h     = (float*)(ws + off);          off += (size_t)BB * DD * 4;
    float*          c     = (float*)(ws + off);          off += (size_t)BB * DD * 4;
    float*          gates = (float*)(ws + off);          off += (size_t)BB * G4 * 4;
    unsigned short* Hbuf  = (unsigned short*)(ws + off); off += (size_t)MROWS * DD * 2;
    float*          maskb = (float*)(ws + off);          off += (size_t)MROWS * 4;
    int*            sind  = (int*)(ws + off);            off += 256;
    int*            declen= (int*)(ws + off);            off += 256;
    (void)off; (void)ws_size; (void)in_sizes; (void)n_in; (void)out_size;

    float* out_tail = out + (size_t)BB * TDEC * VV;  // decode_lengths + sort_ind

    prep_kernel<<<1, 1, 0, stream>>>(lens, sind, declen, out_tail);

    convert_wc_kernel<<<(G4 * KC) / 256, 256, 0, stream>>>(Wih, Whh, Wc);
    bias_sum_kernel<<<G4 / 256, 256, 0, stream>>>(bih, bhh, bsum);
    convert_fcw_kernel<<<(int)(((size_t)VV * DD) / 256), 256, 0, stream>>>(fcW, fcWb);
    zero_hc_kernel<<<(BB * DD) / 256, 256, 0, stream>>>(h, c);

    for (int step = 0; step < 32; ++step) {
        gates_gemm_kernel<<<64, 256, 0, stream>>>(enc, emb, caps, Wc, bsum, h,
                                                  gates, sind, step);
        lstm_update_kernel<<<(BB * DD) / 256, 256, 0, stream>>>(h, c, gates, declen,
                                                                Hbuf, maskb, step);
    }

    fc_gemm_kernel<<<62 * 125, 256, 0, stream>>>(Hbuf, fcWb, fcb, maskb, out);
}
